// TWSM_59700045414683
// MI455X (gfx1250) — compile-verified, matrix operand order checked by reading, not run on hardware
//
#include <hip/hip_runtime.h>
#include <hip/hip_bf16.h>
#include <cstdint>
#include <cstddef>

// ---------------------------------------------------------------------------
// MI455X / gfx1250 implementation of the TWSM reference.
// All heavy math is routed through v_wmma_f32_16x16x32_bf16 (bf16 in, f32 acc).
// Global->LDS staging uses the gfx1250 async-to-LDS path when the toolchain
// exposes the builtin, otherwise plain b128 copies (compile-portable).
// ---------------------------------------------------------------------------

typedef __bf16 bf16;
typedef __attribute__((ext_vector_type(16))) __bf16 v16bf;
typedef __attribute__((ext_vector_type(8)))  __bf16 v8bf;
typedef __attribute__((ext_vector_type(8)))  float  v8f;
typedef __attribute__((ext_vector_type(4)))  int    v4i;

#define AS1 __attribute__((address_space(1)))
#define AS3 __attribute__((address_space(3)))

#ifndef __has_builtin
#define __has_builtin(x) 0
#endif
#if __has_builtin(__builtin_amdgcn_global_load_async_to_lds_b128) && \
    __has_builtin(__builtin_amdgcn_s_wait_asynccnt)
#define HAVE_ASYNC_LDS 1
#else
#define HAVE_ASYNC_LDS 0
#endif

// Geometry (fixed by the reference): 4 clips * 12 frames * 16x16 patches.
#define TOK    12288   // total tokens
#define SEQL   768     // mamba sequence length (3*16*16)
#define NBATCH 16      // mamba batch (4 * 12/3)

__device__ __forceinline__ float siluf(float x)    { return x / (1.f + __expf(-x)); }
__device__ __forceinline__ float geluf(float x)    { return 0.5f * x * (1.f + erff(x * 0.70710678118f)); }
__device__ __forceinline__ float softplusf(float x){ return (x > 20.f) ? x : log1pf(__expf(x)); }

// ---------------------------------------------------------------------------
// fp32 -> bf16 conversion
// ---------------------------------------------------------------------------
__global__ void k_f32_to_bf16(const float* __restrict__ s, bf16* __restrict__ d, long n)
{
    long i = blockIdx.x * 256L + threadIdx.x;
    if (i < n) d[i] = (bf16)s[i];
}

// ---------------------------------------------------------------------------
// WMMA GEMM:  C[M,N] = act( A[M,K] * W[N,K]^T + bias )
// BM=BN=128, BK=32, 256 threads (8 wave32), wave grid 2x4, 4x2 frags/wave.
// ---------------------------------------------------------------------------
#define GBM 128
#define GBN 128
#define GBK 32
#define GST 40   // padded LDS row stride (bf16 elems): conflict-free b128 reads

__global__ __launch_bounds__(256)
void k_gemm_bf16(const bf16* __restrict__ A, const bf16* __restrict__ W,
                 const float* __restrict__ bias, float* __restrict__ Cf,
                 bf16* __restrict__ Cb, int M, int N, int K, int act)
{
    __shared__ __align__(16) bf16 sA[2][GBM * GST];
    __shared__ __align__(16) bf16 sB[2][GBN * GST];

    const int tid  = threadIdx.x;
    const int bm   = blockIdx.x * GBM;
    const int bn   = blockIdx.y * GBN;
    const int wave = tid >> 5;
    const int lane = tid & 31;
    const int wr   = wave >> 2;   // 0..1 : 64 rows per wave
    const int wc   = wave & 3;    // 0..3 : 32 cols per wave
    const int half = lane >> 4;   // 0..1
    const int l16  = lane & 15;

    const int nk = (K + GBK - 1) / GBK;

    v8f zero8 = {0.f,0.f,0.f,0.f,0.f,0.f,0.f,0.f};
    v8f acc[4][2];
#pragma unroll
    for (int i = 0; i < 4; ++i)
#pragma unroll
        for (int j = 0; j < 2; ++j) acc[i][j] = zero8;

    auto stage = [&](int kt, int buf) {
        const int k0 = kt * GBK;
#pragma unroll
        for (int p = 0; p < 2; ++p) {
            const int li  = tid + p * 256;   // 0..511
            const int row = li >> 2;         // 0..127
            const int ch  = (li & 3) << 3;   // 0,8,16,24
            const int gk  = k0 + ch;
            {   // A tile
                bf16* d = &sA[buf][row * GST + ch];
                const int gm = bm + row;
                if (gm < M && (gk + 8) <= K) {
                    const bf16* s = A + (size_t)gm * K + gk;
#if HAVE_ASYNC_LDS
                    __builtin_amdgcn_global_load_async_to_lds_b128((AS1 v4i*)s, (AS3 v4i*)d, 0, 0);
#else
                    *(uint4*)d = *(const uint4*)s;
#endif
                } else {
#pragma unroll
                    for (int j = 0; j < 8; ++j)
                        d[j] = (gm < M && (gk + j) < K) ? A[(size_t)gm * K + gk + j] : (bf16)0.f;
                }
            }
            {   // B tile (weights, row n = output column n)
                bf16* d = &sB[buf][row * GST + ch];
                const int gn = bn + row;
                if (gn < N && (gk + 8) <= K) {
                    const bf16* s = W + (size_t)gn * K + gk;
#if HAVE_ASYNC_LDS
                    __builtin_amdgcn_global_load_async_to_lds_b128((AS1 v4i*)s, (AS3 v4i*)d, 0, 0);
#else
                    *(uint4*)d = *(const uint4*)s;
#endif
                } else {
#pragma unroll
                    for (int j = 0; j < 8; ++j)
                        d[j] = (gn < N && (gk + j) < K) ? W[(size_t)gn * K + gk + j] : (bf16)0.f;
                }
            }
        }
    };

    // ISA 16-bit A/B fragment layout: lanes 0-15 hold K[0..8)+[16..24),
    // lanes 16-31 hold K[8..16)+[24..32) for row (l16).
    auto ldfrag = [&](const bf16* base, int row) -> v16bf {
        const v8bf lo = *(const v8bf*)(base + row * GST + (half << 3));
        const v8bf hi = *(const v8bf*)(base + row * GST + 16 + (half << 3));
        v16bf r;
#pragma unroll
        for (int i = 0; i < 8; ++i) { r[i] = lo[i]; r[i + 8] = hi[i]; }
        return r;
    };

    stage(0, 0);
    for (int kt = 0; kt < nk; ++kt) {
        const int buf = kt & 1;
#if HAVE_ASYNC_LDS
        __builtin_amdgcn_s_wait_asynccnt(0);
#endif
        __syncthreads();
        if (kt + 1 < nk) stage(kt + 1, buf ^ 1);

        v16bf afr[4], bfr[2];
#pragma unroll
        for (int fm = 0; fm < 4; ++fm) afr[fm] = ldfrag(sA[buf], wr * 64 + fm * 16 + l16);
#pragma unroll
        for (int fn = 0; fn < 2; ++fn) bfr[fn] = ldfrag(sB[buf], wc * 32 + fn * 16 + l16);
#pragma unroll
        for (int fm = 0; fm < 4; ++fm)
#pragma unroll
            for (int fn = 0; fn < 2; ++fn)
                acc[fm][fn] = __builtin_amdgcn_wmma_f32_16x16x32_bf16(
                    false, afr[fm], false, bfr[fn], (short)0, acc[fm][fn], false, false);
    }

    // Epilogue. C/D layout: lane = N (l16), VGPR v -> M = v + 8*half.
#pragma unroll
    for (int fn = 0; fn < 2; ++fn) {
        const int gn = bn + wc * 32 + fn * 16 + l16;
        if (gn >= N) continue;
        const float bv = bias ? bias[gn] : 0.f;
#pragma unroll
        for (int fm = 0; fm < 4; ++fm) {
#pragma unroll
            for (int v = 0; v < 8; ++v) {
                const int gm = bm + wr * 64 + fm * 16 + half * 8 + v;
                if (gm >= M) continue;
                float val = acc[fm][fn][v] + bv;
                if (act == 1) val = geluf(val);
                if (Cf) Cf[(size_t)gm * N + gn] = val;
                if (Cb) Cb[(size_t)gm * N + gn] = (bf16)val;
            }
        }
    }
}

// ---------------------------------------------------------------------------
// LayerNorm over last dim C (256 or 512); one block of 256 threads per row.
// Optional fp32 and bf16 outputs.
// ---------------------------------------------------------------------------
__global__ void k_layernorm(const float* __restrict__ x, const float* __restrict__ g,
                            const float* __restrict__ b, float* __restrict__ of,
                            bf16* __restrict__ ob, int C)
{
    const long row = blockIdx.x;
    const float* xr = x + row * (long)C;
    __shared__ float red[2][8];

    float s = 0.f, ss = 0.f;
    for (int c = threadIdx.x; c < C; c += 256) { float v = xr[c]; s += v; ss += v * v; }
#pragma unroll
    for (int off = 16; off > 0; off >>= 1) {
        s  += __shfl_down(s,  off, 32);
        ss += __shfl_down(ss, off, 32);
    }
    const int wave = threadIdx.x >> 5, lane = threadIdx.x & 31;
    if (lane == 0) { red[0][wave] = s; red[1][wave] = ss; }
    __syncthreads();
    if (threadIdx.x == 0) {
        float ts = 0.f, tss = 0.f;
        for (int i = 0; i < 8; ++i) { ts += red[0][i]; tss += red[1][i]; }
        const float m = ts / C;
        const float v = tss / C - m * m;
        red[0][0] = m;
        red[1][0] = rsqrtf(v + 1e-5f);
    }
    __syncthreads();
    const float m = red[0][0], inv = red[1][0];
    for (int c = threadIdx.x; c < C; c += 256) {
        const float v = (xr[c] - m) * inv * g[c] + b[c];
        if (of) of[row * (long)C + c] = v;
        if (ob) ob[row * (long)C + c] = (bf16)v;
    }
}

// ---------------------------------------------------------------------------
// Direction remap. reverse (t -> L-1-t) and h<->w swap commute, so one map
// serves both the input gather and the output accumulate.
// ---------------------------------------------------------------------------
__device__ __forceinline__ int seq_map(int t, int perm, int rev)
{
    int t1 = rev ? (SEQL - 1 - t) : t;
    if (perm) {
        const int s = t1 >> 8, i = (t1 >> 4) & 15, j = t1 & 15;
        t1 = (s << 8) | (j << 4) | i;
    }
    return t1;
}

__global__ void k_gather_bf16(const bf16* __restrict__ src, bf16* __restrict__ dst,
                              int C, int perm, int rev)
{
    long idx = blockIdx.x * 256L + threadIdx.x;
    const long total = (long)TOK * C;
    if (idx >= total) return;
    const int  c   = (int)(idx % C);
    const long row = idx / C;
    const int  t   = (int)(row % SEQL);
    const long b   = row / SEQL;
    dst[idx] = src[(b * SEQL + seq_map(t, perm, rev)) * (long)C + c];
}

__global__ void k_acc_add(float* __restrict__ acc, const float* __restrict__ src,
                          int C, int perm, int rev)
{
    long idx = blockIdx.x * 256L + threadIdx.x;
    const long total = (long)TOK * C;
    if (idx >= total) return;
    const int  c   = (int)(idx % C);
    const long row = idx / C;
    const int  t   = (int)(row % SEQL);
    const long b   = row / SEQL;
    acc[idx] += src[(b * SEQL + seq_map(t, perm, rev)) * (long)C + c];
}

// ---------------------------------------------------------------------------
// Causal depthwise conv (k=4) along L within each batch, + SiLU.
// xz is [TOK, 2*di] (u = first di columns).
// ---------------------------------------------------------------------------
__global__ void k_conv_silu(const float* __restrict__ xz, const float* __restrict__ cw,
                            const float* __restrict__ cb, float* __restrict__ uact,
                            bf16* __restrict__ ubf, int di)
{
    long idx = blockIdx.x * 256L + threadIdx.x;
    const long total = (long)TOK * di;
    if (idx >= total) return;
    const int  d    = (int)(idx % di);
    const long row  = idx / di;
    const int  t    = (int)(row % SEQL);
    const long base = (row - t) * (long)(2 * di) + d;
    float s = cb[d];
#pragma unroll
    for (int k = 0; k < 4; ++k) {
        const int ts = t + k - 3;
        if (ts >= 0) s += xz[base + (long)ts * (2 * di)] * cw[d * 4 + k];
    }
    const float v = siluf(s);
    uact[idx] = v;
    ubf[idx]  = (bf16)v;
}

// ---------------------------------------------------------------------------
// Strided slice -> bf16 (dt columns of xdb).
// ---------------------------------------------------------------------------
__global__ void k_slice_bf16(const float* __restrict__ src, bf16* __restrict__ dst,
                             int ld, int n, long M)
{
    long idx = blockIdx.x * 256L + threadIdx.x;
    const long total = M * n;
    if (idx >= total) return;
    const int  c   = (int)(idx % n);
    const long row = idx / n;
    dst[idx] = (bf16)src[row * (long)ld + c];
}

// ---------------------------------------------------------------------------
// Fused softplus + selective scan (N=1) + D skip + SiLU gate -> bf16.
// One thread per (batch, channel); sequential over L=768.
// ---------------------------------------------------------------------------
__global__ void k_scan(const float* __restrict__ xz, const float* __restrict__ uact,
                       const float* __restrict__ xdb, const float* __restrict__ dtlin,
                       const float* __restrict__ dtb, const float* __restrict__ A_log,
                       const float* __restrict__ Dp, bf16* __restrict__ ybf,
                       int di, int R, int Nx)
{
    const int idx = blockIdx.x * 256 + threadIdx.x;
    if (idx >= NBATCH * di) return;
    const int d = idx % di;
    const int b = idx / di;
    const float Ad   = -__expf(A_log[d]);
    const float Dd   = Dp[d];
    const float bias = dtb[d];
    const long  r0   = (long)b * SEQL;
    float h = 0.f;
    for (int t = 0; t < SEQL; ++t) {
        const long row = r0 + t;
        __builtin_prefetch(&dtlin[(row + 8) * (long)di + d], 0, 1);  // global_prefetch_b8
        __builtin_prefetch(&uact [(row + 8) * (long)di + d], 0, 1);
        const float dt = softplusf(dtlin[row * (long)di + d] + bias);
        const float u  = uact[row * (long)di + d];
        const float Bt = xdb[row * (long)Nx + R];
        const float Ct = xdb[row * (long)Nx + R + 1];
        h = __expf(dt * Ad) * h + dt * Bt * u;
        const float z = xz[row * (long)(2 * di) + di + d];
        const float y = (h * Ct + u * Dd) * siluf(z);
        ybf[row * (long)di + d] = (bf16)y;
    }
}

// ---------------------------------------------------------------------------
// _switch: roll first quarter of channels by +shift and last quarter by -shift
// along S=12. Gather form (src != dst).
// ---------------------------------------------------------------------------
__global__ void k_switch(const float* __restrict__ src, float* __restrict__ dst,
                         int C, int shift)
{
    long idx = blockIdx.x * 256L + threadIdx.x;
    const long total = (long)TOK * C;
    if (idx >= total) return;
    const int  c   = (int)(idx % C);
    const long row = idx / C;
    const int  p   = (int)(row % 256);
    const long r2  = row / 256;
    const int  s   = (int)(r2 % 12);
    const int  n   = (int)(r2 / 12);
    const int  wl  = C >> 2;
    int ss = s;
    if (c < wl)           ss = ((s - shift) % 12 + 12) % 12;
    else if (c >= 3 * wl) ss = ((s + shift) % 12 + 12) % 12;
    dst[idx] = src[(((long)n * 12 + ss) * 256 + p) * (long)C + c];
}

// ---------------------------------------------------------------------------
// Patch-conv im2col: x (4,256,12,32,32) -> A [12288, 1024] bf16,
// feature f = c*4 + dy*2 + dx to match patch_w (E,256,1,2,2) flattening.
// ---------------------------------------------------------------------------
__global__ void k_patch_gather(const float* __restrict__ x, bf16* __restrict__ A)
{
    long idx = blockIdx.x * 256L + threadIdx.x;
    const long total = (long)TOK * 1024;
    if (idx >= total) return;
    const int  f   = (int)(idx % 1024);
    const long tk  = idx / 1024;
    const int  c   = f >> 2;
    const int  dy  = (f >> 1) & 1;
    const int  dx  = f & 1;
    const int  w   = (int)(tk % 16);
    const int  h   = (int)((tk / 16) % 16);
    const int  d   = (int)((tk / 256) % 12);
    const int  n   = (int)(tk / 3072);
    A[idx] = (bf16)x[((((long)n * 256 + c) * 12 + d) * 32 + (h * 2 + dy)) * 32 + (w * 2 + dx)];
}

// ---------------------------------------------------------------------------
// Final NDHWC(4,12,16,16,512) -> NCDHW(4,512,12,16,16)
// ---------------------------------------------------------------------------
__global__ void k_out_transpose(const float* __restrict__ src, float* __restrict__ dst)
{
    long idx = blockIdx.x * 256L + threadIdx.x;
    const long total = (long)TOK * 512;
    if (idx >= total) return;
    const int  w = (int)(idx & 15);
    const int  h = (int)((idx >> 4) & 15);
    const int  d = (int)((idx >> 8) % 12);
    const long r = idx / (256L * 12);
    const int  c = (int)(r % 512);
    const int  n = (int)(r / 512);
    dst[idx] = src[(((long)n * 12 + d) * 256 + h * 16 + w) * 512L + c];
}

__global__ void k_zero(float* __restrict__ d, long n)
{
    long i = blockIdx.x * 256L + threadIdx.x;
    if (i < n) d[i] = 0.f;
}

__global__ void k_add(float* __restrict__ dst, const float* __restrict__ src, long n)
{
    long i = blockIdx.x * 256L + threadIdx.x;
    if (i < n) dst[i] += src[i];
}

// ---------------------------------------------------------------------------
// Host orchestration
// ---------------------------------------------------------------------------
struct MambaP {
    const float *in_proj_w, *conv_w, *conv_b, *x_proj_w, *dt_proj_w,
                *dt_proj_b, *A_log, *Dp, *out_proj_w;
};
struct BlockP {
    MambaP m;
    const float *n1_g, *n1_b, *n2_g, *n2_b, *fc1_w, *fc1_b, *fc2_w, *fc2_b;
};

extern "C" void kernel_launch(void* const* d_in, const int* in_sizes, int n_in,
                              void* d_out, int out_size, void* d_ws, size_t ws_size,
                              hipStream_t stream)
{
    (void)in_sizes; (void)n_in; (void)out_size; (void)ws_size;

    // --- unpack inputs in setup_inputs() dict order ---
    int ip = 0;
    auto nxt = [&]() { return (const float*)d_in[ip++]; };
    auto rdMamba = [&]() {
        MambaP m;
        m.in_proj_w = nxt(); m.conv_w = nxt(); m.conv_b = nxt();
        m.x_proj_w = nxt();  m.dt_proj_w = nxt(); m.dt_proj_b = nxt();
        m.A_log = nxt();     m.Dp = nxt();        m.out_proj_w = nxt();
        return m;
    };
    auto rdBlock = [&]() {
        BlockP b;
        b.m = rdMamba();
        b.n1_g = nxt(); b.n1_b = nxt(); b.n2_g = nxt(); b.n2_b = nxt();
        b.fc1_w = nxt(); b.fc1_b = nxt(); b.fc2_w = nxt(); b.fc2_b = nxt();
        return b;
    };
    const float* x       = nxt();
    const float* patch_w = nxt();
    const float* patch_g = nxt();
    const float* patch_b = nxt();
    BlockP B1 = rdBlock(), B2 = rdBlock(), B3 = rdBlock(), B4 = rdBlock();
    const float* mid_g    = nxt();
    const float* mid_b    = nxt();
    const float* mid_w    = nxt();
    const float* mid_bias = nxt();
    BlockP B5 = rdBlock(), B6 = rdBlock();

    // --- workspace arena (sized for worst case C=512, di=1024) ---
    char* wp = (char*)d_ws;
    auto alloc = [&](size_t bytes) { void* p = (void*)wp; wp += (bytes + 255) & ~(size_t)255; return p; };
    float* cur   = (float*)alloc((size_t)TOK * 512 * 4);
    float* aux   = (float*)alloc((size_t)TOK * 512 * 4);
    float* acc   = (float*)alloc((size_t)TOK * 512 * 4);
    float* mout  = (float*)alloc((size_t)TOK * 512 * 4);
    bf16*  lnb   = (bf16*) alloc((size_t)TOK * 512 * 2);
    bf16*  xseq  = (bf16*) alloc((size_t)TOK * 512 * 2);
    float* xz    = (float*)alloc((size_t)TOK * 2048 * 4);
    float* uact  = (float*)alloc((size_t)TOK * 1024 * 4);
    bf16*  ubf   = (bf16*) alloc((size_t)TOK * 1024 * 2);
    float* xdb   = (float*)alloc((size_t)TOK * 34 * 4);
    bf16*  dtin  = (bf16*) alloc((size_t)TOK * 32 * 2);
    float* dtlin = (float*)alloc((size_t)TOK * 1024 * 4);
    bf16*  ybf   = (bf16*) alloc((size_t)TOK * 1024 * 2);
    bf16*  hidbf = (bf16*) alloc((size_t)TOK * 2048 * 2);
    bf16*  w_in  = (bf16*) alloc((size_t)2048 * 512 * 2);
    bf16*  w_x   = (bf16*) alloc((size_t)34 * 1024 * 2);
    bf16*  w_dt  = (bf16*) alloc((size_t)1024 * 32 * 2);
    bf16*  w_out = (bf16*) alloc((size_t)512 * 1024 * 2);
    bf16*  w_fc1 = (bf16*) alloc((size_t)2048 * 512 * 2);
    bf16*  w_fc2 = (bf16*) alloc((size_t)512 * 2048 * 2);

    auto ew = [](long n) { return dim3((unsigned)((n + 255) / 256)); };
    auto cvt = [&](const float* s, bf16* d, long n) {
        k_f32_to_bf16<<<ew(n), 256, 0, stream>>>(s, d, n);
    };
    auto gemm = [&](const bf16* Ai, const bf16* Wi, const float* bi, float* Co,
                    bf16* Cbo, int M, int N, int K, int act) {
        dim3 g((M + GBM - 1) / GBM, (N + GBN - 1) / GBN);
        k_gemm_bf16<<<g, 256, 0, stream>>>(Ai, Wi, bi, Co, Cbo, M, N, K, act);
    };

    auto run_block = [&](const BlockP& p, float* curp, int C) {
        const int di = 2 * C;
        const int R  = (C + 15) / 16;   // dt_rank: 16 @C=256, 32 @C=512
        const int Nx = R + 2;

        cvt(p.m.in_proj_w,  w_in,  (long)2 * di * C);
        cvt(p.m.x_proj_w,   w_x,   (long)Nx * di);
        cvt(p.m.dt_proj_w,  w_dt,  (long)di * R);
        cvt(p.m.out_proj_w, w_out, (long)C * di);
        cvt(p.fc1_w, w_fc1, (long)4 * C * C);
        cvt(p.fc2_w, w_fc2, (long)4 * C * C);

        k_layernorm<<<TOK, 256, 0, stream>>>(curp, p.n1_g, p.n1_b, (float*)nullptr, lnb, C);
        k_zero<<<ew((long)TOK * C), 256, 0, stream>>>(acc, (long)TOK * C);

        for (int dir = 0; dir < 4; ++dir) {
            const int perm = (dir >= 2), rev = (dir & 1);
            k_gather_bf16<<<ew((long)TOK * C), 256, 0, stream>>>(lnb, xseq, C, perm, rev);
            gemm(xseq, w_in, nullptr, xz, nullptr, TOK, 2 * di, C, 0);
            k_conv_silu<<<ew((long)TOK * di), 256, 0, stream>>>(xz, p.m.conv_w, p.m.conv_b, uact, ubf, di);
            gemm(ubf, w_x, nullptr, xdb, nullptr, TOK, Nx, di, 0);
            k_slice_bf16<<<ew((long)TOK * R), 256, 0, stream>>>(xdb, dtin, Nx, R, (long)TOK);
            gemm(dtin, w_dt, nullptr, dtlin, nullptr, TOK, di, R, 0);
            k_scan<<<(NBATCH * di + 255) / 256, 256, 0, stream>>>(
                xz, uact, xdb, dtlin, p.m.dt_proj_b, p.m.A_log, p.m.Dp, ybf, di, R, Nx);
            gemm(ybf, w_out, nullptr, mout, nullptr, TOK, C, di, 0);
            k_acc_add<<<ew((long)TOK * C), 256, 0, stream>>>(acc, mout, C, perm, rev);
        }
        k_add<<<ew((long)TOK * C), 256, 0, stream>>>(curp, acc, (long)TOK * C);

        k_layernorm<<<TOK, 256, 0, stream>>>(curp, p.n2_g, p.n2_b, (float*)nullptr, lnb, C);
        gemm(lnb, w_fc1, p.fc1_b, nullptr, hidbf, TOK, 4 * C, C, 1 /*gelu*/);
        gemm(hidbf, w_fc2, p.fc2_b, mout, nullptr, TOK, C, 4 * C, 0);
        k_add<<<ew((long)TOK * C), 256, 0, stream>>>(curp, mout, (long)TOK * C);
    };

    auto do_switch = [&](int shift, int C) {
        k_switch<<<ew((long)TOK * C), 256, 0, stream>>>(cur, aux, C, shift);
        float* t = cur; cur = aux; aux = t;
    };

    // --- patch embedding: im2col + GEMM + LayerNorm ---
    k_patch_gather<<<ew((long)TOK * 1024), 256, 0, stream>>>(x, ubf);
    cvt(patch_w, w_fc1, 256L * 1024);
    gemm(ubf, w_fc1, nullptr, aux, nullptr, TOK, 256, 1024, 0);
    k_layernorm<<<TOK, 256, 0, stream>>>(aux, patch_g, patch_b, cur, (bf16*)nullptr, 256);

    // --- stage 1: four blocks @ C=256 ---
    do_switch(1, 256);  run_block(B1, cur, 256);
    do_switch(-2, 256); run_block(B2, cur, 256);
    do_switch(2, 256);  run_block(B3, cur, 256);
    do_switch(-2, 256); run_block(B4, cur, 256);
    do_switch(1, 256);

    // --- mid projection 256 -> 512 ---
    k_layernorm<<<TOK, 256, 0, stream>>>(cur, mid_g, mid_b, (float*)nullptr, lnb, 256);
    cvt(mid_w, w_in, 512L * 256);
    gemm(lnb, w_in, mid_bias, aux, nullptr, TOK, 512, 256, 0);
    { float* t = cur; cur = aux; aux = t; }

    // --- stage 2: two blocks @ C=512 ---
    do_switch(1, 512);  run_block(B5, cur, 512);
    do_switch(-2, 512); run_block(B6, cur, 512);
    do_switch(1, 512);

    // --- final transpose to NCDHW output ---
    k_out_transpose<<<ew((long)TOK * 512), 256, 0, stream>>>(cur, (float*)d_out);
}
